// SiblingNDMV_16192026706013
// MI455X (gfx1250) — matrix operand order repaired and checked by reading, not compiled, not run on hardware
//
#include <hip/hip_runtime.h>

// ---------- types ----------
typedef __attribute__((ext_vector_type(16))) __bf16 v16bf;
typedef __attribute__((ext_vector_type(8)))  float  v8f;
typedef __attribute__((ext_vector_type(4)))  unsigned int v4u;
typedef __attribute__((ext_vector_type(8)))  int v8i;
typedef __attribute__((ext_vector_type(4)))  int v4i;

union ABfrag { v16bf v; unsigned int u[8]; };

#ifndef __has_builtin
#define __has_builtin(x) 0
#endif
#if __has_builtin(__builtin_amdgcn_tensor_load_to_lds)
#define HAVE_TDM 1
#else
#define HAVE_TDM 0
#endif

__device__ __forceinline__ float b2f(unsigned short s) {
    unsigned int u = ((unsigned int)s) << 16;
    float f; __builtin_memcpy(&f, &u, 4); return f;
}
__device__ __forceinline__ unsigned short f2b(float f) {
    unsigned int u; __builtin_memcpy(&u, &f, 4);
    unsigned int r = (u + 0x7FFFu + ((u >> 16) & 1u)) >> 16;
    return (unsigned short)r;
}

#if HAVE_TDM
// 2D TDM load: bf16 tile [tile_d1 rows x 32 elems], row stride = stride_elems,
// into LDS with pad_enable (pad_interval=16 DWORDs, pad_amount code = 1 or 2 DW).
// OOB rows/cols (beyond tensor_d0/d1) fill LDS with zeros.
__device__ __forceinline__ void tdm_load_2d(unsigned int lds_off, const void* gaddr,
                                            int tile_d0, int tile_d1,
                                            int tensor_d0, int tensor_d1,
                                            int stride_elems, int pad_amount_code)
{
    unsigned long long ga = (unsigned long long)gaddr;
    v4u g0;
    g0[0] = 1u;                                     // count=1, user descriptor
    g0[1] = lds_off;                                // lds_addr (bytes)
    g0[2] = (unsigned int)(ga & 0xffffffffu);       // global_addr[31:0]
    g0[3] = (unsigned int)((ga >> 32) & 0x01ffffffu) | (2u << 30); // addr[56:32], type=2
    v8i g1;
    g1[0] = (int)((1u << 16)                        // data_size = 2 bytes
                | (1u << 20)                        // pad_enable
                | (3u << 22)                        // pad_interval: 16 DWORDs (64B rows)
                | ((unsigned int)pad_amount_code << 25));
    g1[1] = (int)((unsigned int)tensor_d0 << 16);   // tensor_dim0[15:0] @ bits63:48
    g1[2] = (int)(((unsigned int)tensor_d0 >> 16) | ((unsigned int)tensor_d1 << 16));
    g1[3] = (int)((((unsigned int)tensor_d1 >> 16) & 0xffffu) | ((unsigned int)tile_d0 << 16));
    g1[4] = tile_d1;                                // tile_dim1 ; tile_dim2=0
    g1[5] = stride_elems;                           // tensor_dim0_stride[31:0]
    g1[6] = 0;
    g1[7] = 0;
    v4i gz = {0, 0, 0, 0};
#if __clang_major__ >= 23
    v8i gz8 = {0, 0, 0, 0, 0, 0, 0, 0};
    __builtin_amdgcn_tensor_load_to_lds(g0, g1, gz, gz, gz8, 0);
#else
    __builtin_amdgcn_tensor_load_to_lds(g0, g1, gz, gz, 0);
#endif
}
#endif

// =====================================================================
// Tiled bf16 WMMA GEMM:  out = post( A[M,K] @ W[K,N] )
//  - A: bf16 row-major, lda=K ; WT: bf16 TRANSPOSED weights [N][K]
//  - optional bias[N] (f32), optional residual-relu: out = res + relu(acc+bias)
//  - optional output row remap (sibling: r -> (r/64)*65 + r%64 + 1)
//  - blockIdx.z = (d,v) plane via element strides
//  - staging via Tensor Data Mover (TENSORcnt) when available
// =====================================================================
__global__ __launch_bounds__(128) void k_gemm(
    const unsigned short* __restrict__ A, long long a_dvs,
    const unsigned short* __restrict__ WT, long long w_dvs,
    const float* __restrict__ bias, long long b_dvs,
    const unsigned short* __restrict__ res, long long r_dvs,
    void* __restrict__ Ovoid, int o_bf16, long long o_dvs,
    int M, int N, int K, int relu_res, int row_remap)
{
    __shared__ unsigned short As[64 * 36];   // [row][k], 16DW data + 2DW pad
    __shared__ unsigned short Bs[64 * 34];   // [col][k], 16DW data + 1DW pad

    const int tid = threadIdx.x;
    const int dv  = blockIdx.z;
    const int n0  = blockIdx.x * 64;
    const int m0  = blockIdx.y * 64;

    const unsigned short* Ab = A  + (long long)dv * a_dvs;
    const unsigned short* Wt = WT + (long long)dv * w_dvs;

    const int wid = tid >> 5, lane = tid & 31, hf = lane >> 4, mr = lane & 15;

    v8f zero = {0.f,0.f,0.f,0.f,0.f,0.f,0.f,0.f};
    v8f acc[4]; acc[0]=zero; acc[1]=zero; acc[2]=zero; acc[3]=zero;

    for (int kb = 0; kb < K; kb += 32) {
        __syncthreads();
#if HAVE_TDM
        if (wid == 0) {
            tdm_load_2d((unsigned int)(unsigned long long)(const void*)As,
                        Ab + (long long)m0 * K + kb, 32, 64, K - kb, M - m0, K, 1);
            tdm_load_2d((unsigned int)(unsigned long long)(const void*)Bs,
                        Wt + (long long)n0 * K + kb, 32, 64, K - kb, N - n0, K, 0);
            __builtin_amdgcn_s_wait_tensorcnt((short)0);
        }
#else
        for (int idx = tid; idx < 64 * 32; idx += 128) {
            int r = idx >> 5, k = idx & 31;
            int gr = m0 + r;
            As[r * 36 + k] = (gr < M) ? Ab[(long long)gr * K + kb + k] : (unsigned short)0;
        }
        for (int idx = tid; idx < 64 * 32; idx += 128) {
            int c = idx >> 5, k = idx & 31;
            int gc = n0 + c;
            Bs[c * 34 + k] = (gc < N) ? Wt[(long long)gc * K + kb + k] : (unsigned short)0;
        }
#endif
        __syncthreads();

        ABfrag a;
        const unsigned short* ap = As + (wid * 16 + mr) * 36;
#pragma unroll
        for (int j = 0; j < 8; ++j) {
            int k0 = (j < 4) ? (2 * j + hf * 8) : (2 * j + 8 + hf * 8);
            a.u[j] = *(const unsigned int*)(ap + k0);
        }
#pragma unroll
        for (int ct = 0; ct < 4; ++ct) {
            ABfrag bb;
            const unsigned short* bp = Bs + (ct * 16 + mr) * 34;
#pragma unroll
            for (int j = 0; j < 8; ++j)
                bb.u[j] = *(const unsigned int*)(bp + ((j < 4) ? (2 * j + hf * 8) : (2 * j + 8 + hf * 8)));
            acc[ct] = __builtin_amdgcn_wmma_f32_16x16x32_bf16(
                false, a.v, false, bb.v, (short)0, acc[ct], false, false);
        }
    }

    const float* biasp = bias ? bias + (long long)dv * b_dvs : nullptr;
    const unsigned short* resp = res ? res + (long long)dv * r_dvs : nullptr;
    unsigned short* Ob = (unsigned short*)Ovoid + (long long)dv * o_dvs;
    float*          Of = (float*)Ovoid + (long long)dv * o_dvs;

#pragma unroll
    for (int ct = 0; ct < 4; ++ct) {
        int gc = n0 + ct * 16 + mr;
        if (gc >= N) continue;
#pragma unroll
        for (int p = 0; p < 8; ++p) {
            int gr = m0 + wid * 16 + p + hf * 8;
            if (gr >= M) continue;
            float v = acc[ct][p];
            if (biasp) v += biasp[gc];
            if (relu_res) v = fmaxf(v, 0.f) + b2f(resp[(long long)gr * N + gc]);
            long long orow = gr;
            if (row_remap) orow = (long long)(gr >> 6) * 65 + (gr & 63) + 1;
            long long oidx = orow * (long long)N + gc;
            if (o_bf16) Ob[oidx] = f2b(v); else Of[oidx] = v;
        }
    }
}

// ---------- prep kernels ----------
__global__ void k_cvt(const float* __restrict__ src, unsigned short* __restrict__ dst, int n) {
    for (int i = blockIdx.x * blockDim.x + threadIdx.x; i < n; i += gridDim.x * blockDim.x)
        dst[i] = f2b(src[i]);
}

// W: [nmat][K][N] f32  ->  WT: [nmat][N][K] bf16
__global__ void k_cvt_t(const float* __restrict__ W, unsigned short* __restrict__ WT,
                        int K, int N, int nmat) {
    long long total = (long long)nmat * K * N;
    for (long long idx = blockIdx.x * (long long)blockDim.x + threadIdx.x; idx < total;
         idx += (long long)gridDim.x * blockDim.x) {
        long long m = idx / ((long long)K * N);
        long long rem = idx - m * (long long)K * N;
        int n = (int)(rem % N), k = (int)(rem / N);
        WT[(m * N + n) * K + k] = f2b(W[idx]);
    }
}

__global__ void k_gather(const int* __restrict__ pos_tag,
                         const float* __restrict__ pos_emb,
                         unsigned short* __restrict__ X) {
    int row = blockIdx.x;                      // 4096
    int tag = pos_tag[row];
    const float* src = pos_emb + (long long)tag * 256;
    for (int p = threadIdx.x; p < 256; p += blockDim.x)
        X[(long long)row * 256 + p] = f2b(src[p]);
}

__global__ void k_nosib(const float* __restrict__ pos_emb,
                        unsigned short* __restrict__ Yall) {
    int b = blockIdx.x;                        // 64
    for (int p = threadIdx.x; p < 256; p += blockDim.x)
        Yall[((long long)4096 + (long long)b * 65) * 256 + p] = f2b(pos_emb[35 * 256 + p]);
}

__global__ void k_zero16(unsigned short* __restrict__ p, int n) {
    int i = blockIdx.x * 256 + threadIdx.x;
    if (i < n) p[i] = 0;
}

// =====================================================================
// Fused attach: per (b,i) block
//   S[m,r] = fp[b,i,dv,r]*fs[b,m,dv,r]   (LDS, bf16)
//   R[dv,c,m] = Fc[dv] (48x128) @ S^T    (WMMA, f32 acc)
//   log-softmax over c, left/right mask + child gather, write [j,m,v]
// =====================================================================
__global__ __launch_bounds__(256) void k_attach(
    const unsigned short* __restrict__ FP,   // [4][4096][128] bf16
    const unsigned short* __restrict__ FS,   // [4][4160][128] bf16
    const unsigned short* __restrict__ FC,   // [4][48][128]   bf16 (rows>=34 zero)
    const int* __restrict__ pos_tag,
    float* __restrict__ out)
{
    __shared__ unsigned short Bsh[80 * 130];
    __shared__ float Rs[4 * 34 * 80];

    const int tid = threadIdx.x;
    const int b = blockIdx.x >> 6, i = blockIdx.x & 63;
    const int wid = tid >> 5, lane = tid & 31, hf = lane >> 4, mr = lane & 15;

    for (int dv = 0; dv < 4; ++dv) {
        __syncthreads();
        const unsigned short* fpr = FP + ((long long)dv * 4096 + b * 64 + i) * 128;
        for (int idx = tid; idx < 80 * 128; idx += 256) {
            int m = idx >> 7, r = idx & 127;
            unsigned short v = 0;
            if (m < 65) {
                float p = b2f(fpr[r]) *
                          b2f(FS[((long long)dv * 4160 + b * 65 + m) * 128 + r]);
                v = f2b(p);
            }
            Bsh[m * 130 + r] = v;
        }
        __syncthreads();

        for (int tile = wid; tile < 15; tile += 8) {   // wave-uniform trip count
            int rt = tile / 5, ct = tile % 5;
            v8f acc = {0.f,0.f,0.f,0.f,0.f,0.f,0.f,0.f};
            const unsigned short* arow = FC + ((long long)dv * 48 + rt * 16 + mr) * 128;
            const unsigned short* bcol = Bsh + (ct * 16 + mr) * 130;
#pragma unroll
            for (int ks = 0; ks < 4; ++ks) {
                int kb = ks * 32;
                ABfrag a, bb;
#pragma unroll
                for (int j = 0; j < 8; ++j) {
                    int k0 = (j < 4) ? (2 * j + hf * 8) : (2 * j + 8 + hf * 8);
                    a.u[j] = *(const unsigned int*)(arow + kb + k0);
                }
#pragma unroll
                for (int j = 0; j < 8; ++j)
                    bb.u[j] = *(const unsigned int*)(bcol + kb + ((j < 4) ? (2 * j + hf * 8) : (2 * j + 8 + hf * 8)));
                acc = __builtin_amdgcn_wmma_f32_16x16x32_bf16(
                    false, a.v, false, bb.v, (short)0, acc, false, false);
            }
#pragma unroll
            for (int p = 0; p < 8; ++p) {
                int c = rt * 16 + p + hf * 8;
                int m = ct * 16 + mr;
                if (c < 34) Rs[(dv * 34 + c) * 80 + m] = acc[p];
            }
        }
    }
    __syncthreads();

    for (int idx = tid; idx < 4 * 65; idx += 256) {
        int dv = idx / 65, m = idx % 65;
        float mx = -1e30f;
        for (int c = 0; c < 34; ++c) mx = fmaxf(mx, Rs[(dv * 34 + c) * 80 + m]);
        float s = 0.f;
        for (int c = 0; c < 34; ++c) s += __expf(Rs[(dv * 34 + c) * 80 + m] - mx);
        float lse = mx + __logf(s);
        for (int c = 0; c < 34; ++c) Rs[(dv * 34 + c) * 80 + m] -= lse;
    }
    __syncthreads();

    long long base = (long long)blockIdx.x * 64 * 65 * 2;
    for (int idx = tid; idx < 64 * 65; idx += 256) {
        int j = idx / 65, m = idx % 65;
        float v0 = 0.f, v1 = 0.f;
        if (j != i) {
            int c = pos_tag[b * 64 + j] - 1; if (c < 0) c = 0;
            int d = (j < i) ? 0 : 1;
            v0 = Rs[((d * 2 + 0) * 34 + c) * 80 + m];
            v1 = Rs[((d * 2 + 1) * 34 + c) * 80 + m];
        }
        out[base + (long long)idx * 2 + 0] = v0;
        out[base + (long long)idx * 2 + 1] = v1;
    }
}

// ---------- decision ----------
__global__ __launch_bounds__(256) void k_decision(
    const unsigned short* __restrict__ GP,   // [4][4096][128]
    const unsigned short* __restrict__ GS,   // [4][4160][128]
    const unsigned short* __restrict__ GD,   // [4][2][128]
    float* __restrict__ out)
{
    __shared__ float gpL[4][128];
    __shared__ float gdL[4][2][128];
    const int tid = threadIdx.x;
    const int b = blockIdx.x >> 6;

    for (int idx = tid; idx < 512; idx += 256) {
        int dv = idx >> 7, r = idx & 127;
        gpL[dv][r] = b2f(GP[((long long)dv * 4096 + blockIdx.x) * 128 + r]);
    }
    for (int idx = tid; idx < 1024; idx += 256) {
        int dv = idx >> 8, k = (idx >> 7) & 1, r = idx & 127;
        gdL[dv][k][r] = b2f(GD[(dv * 2 + k) * 128 + r]);
    }
    __syncthreads();

    long long base = (long long)blockIdx.x * 65 * 4 * 2;
    for (int idx = tid; idx < 260; idx += 256) {
        int m = idx >> 2, dv = idx & 3;
        const unsigned short* gs = GS + ((long long)dv * 4160 + b * 65 + m) * 128;
        float v0 = 0.f, v1 = 0.f;
        for (int r = 0; r < 128; ++r) {
            float t = gpL[dv][r] * b2f(gs[r]);
            v0 += t * gdL[dv][0][r];
            v1 += t * gdL[dv][1][r];
        }
        float mx = fmaxf(v0, v1);
        float lse = mx + __logf(__expf(v0 - mx) + __expf(v1 - mx));
        long long o = base + ((long long)m * 4 + dv) * 2;
        out[o + 0] = v0 - lse;
        out[o + 1] = v1 - lse;
    }
}

// ---------- root ----------
__global__ __launch_bounds__(256) void k_root(
    const unsigned short* __restrict__ RR,   // [4][64]
    const unsigned short* __restrict__ RC,   // [4][34][64]
    const int* __restrict__ pos_tag,
    float* __restrict__ out)
{
    __shared__ float rv[34];
    __shared__ float lse_s;
    const int tid = threadIdx.x;
    if (tid < 34) {
        float s = 0.f;
        for (int dv = 0; dv < 4; ++dv)
            for (int r = 0; r < 64; ++r)
                s += b2f(RR[dv * 64 + r]) * b2f(RC[(dv * 34 + tid) * 64 + r]);
        rv[tid] = s;
    }
    __syncthreads();
    if (tid == 0) {
        float mx = -1e30f;
        for (int c = 0; c < 34; ++c) mx = fmaxf(mx, rv[c]);
        float s = 0.f;
        for (int c = 0; c < 34; ++c) s += __expf(rv[c] - mx);
        lse_s = mx + __logf(s);
    }
    __syncthreads();
    float l = lse_s;
    for (int idx = tid; idx < 4096; idx += 256) {
        int c = pos_tag[idx] - 1; if (c < 0) c = 0;
        out[idx] = rv[c] - l;
    }
}

// ---------- host ----------
static void launch_gemm(hipStream_t s,
    const unsigned short* A, long long a_dvs,
    const unsigned short* WT, long long w_dvs,
    const float* bias, long long b_dvs,
    const unsigned short* res, long long r_dvs,
    void* O, int o_bf16, long long o_dvs,
    int M, int N, int K, int relu_res, int remap, int dvn)
{
    dim3 grid((N + 63) / 64, (M + 63) / 64, dvn);
    k_gemm<<<grid, 128, 0, s>>>(A, a_dvs, WT, w_dvs, bias, b_dvs,
                                res, r_dvs, O, o_bf16, o_dvs,
                                M, N, K, relu_res, remap);
}

extern "C" void kernel_launch(void* const* d_in, const int* in_sizes, int n_in,
                              void* d_out, int out_size, void* d_ws, size_t ws_size,
                              hipStream_t stream)
{
    (void)in_sizes; (void)n_in; (void)out_size; (void)ws_size;

    const int*   pos_tag  = (const int*)d_in[0];
    const float* pos_emb  = (const float*)d_in[1];
    const float* root_emb = (const float*)d_in[2];
    const float* dec_emb  = (const float*)d_in[3];
    const float* Wp = (const float*)d_in[4],  *bp = (const float*)d_in[5];
    const float* Ws = (const float*)d_in[6],  *bs = (const float*)d_in[7];
    const float* Wc = (const float*)d_in[8],  *bc = (const float*)d_in[9];
    const float* Wr = (const float*)d_in[10], *br = (const float*)d_in[11];
    const float* Wd = (const float*)d_in[12], *bd = (const float*)d_in[13];
    const float* Wenc = (const float*)d_in[14], *benc = (const float*)d_in[15];
    const float* Ua1 = (const float*)d_in[16], *Ua2 = (const float*)d_in[17], *Ua3 = (const float*)d_in[18];
    const float* Ud1 = (const float*)d_in[19], *Ud2 = (const float*)d_in[20], *Ud3 = (const float*)d_in[21];
    const float* Ur1 = (const float*)d_in[22], *Ur2 = (const float*)d_in[23];

    char* ws = (char*)d_ws;
    auto carve = [&](size_t bytes) { char* p = ws; ws += (bytes + 255) & ~(size_t)255; return p; };
    // activations (bf16)
    unsigned short* X    = (unsigned short*)carve(4096ull * 256 * 2);
    unsigned short* Yall = (unsigned short*)carve(8293ull * 256 * 2);   // par 0, sib 4096, child 8256, root 8290, dec 8291
    unsigned short* Hall = (unsigned short*)carve(4ull * 8293 * 256 * 2);
    unsigned short* FPb  = (unsigned short*)carve(4ull * 4096 * 128 * 2);
    unsigned short* GPb  = (unsigned short*)carve(4ull * 4096 * 128 * 2);
    unsigned short* FSb  = (unsigned short*)carve(4ull * 4160 * 128 * 2);
    unsigned short* GSb  = (unsigned short*)carve(4ull * 4160 * 128 * 2);
    unsigned short* FCb  = (unsigned short*)carve(4ull * 48 * 128 * 2);
    unsigned short* GDb  = (unsigned short*)carve(4ull * 2 * 128 * 2);
    unsigned short* RRb  = (unsigned short*)carve(4ull * 64 * 2);
    unsigned short* RCb  = (unsigned short*)carve(4ull * 34 * 64 * 2);
    // converted inputs (bf16)
    unsigned short* PembB = (unsigned short*)carve(36ull * 256 * 2);
    unsigned short* RembB = (unsigned short*)carve(256ull * 2);
    unsigned short* DembB = (unsigned short*)carve(512ull * 2);
    // transposed bf16 weights [N][K]
    unsigned short* WpT  = (unsigned short*)carve(65536ull * 2);
    unsigned short* WsT  = (unsigned short*)carve(65536ull * 2);
    unsigned short* WcT  = (unsigned short*)carve(65536ull * 2);
    unsigned short* WrT  = (unsigned short*)carve(65536ull * 2);
    unsigned short* WdT  = (unsigned short*)carve(65536ull * 2);
    unsigned short* WencT= (unsigned short*)carve(4ull * 65536 * 2);
    unsigned short* Ua1T = (unsigned short*)carve(32768ull * 2);
    unsigned short* Ua2T = (unsigned short*)carve(32768ull * 2);
    unsigned short* Ua3T = (unsigned short*)carve(32768ull * 2);
    unsigned short* Ud1T = (unsigned short*)carve(32768ull * 2);
    unsigned short* Ud2T = (unsigned short*)carve(32768ull * 2);
    unsigned short* Ud3T = (unsigned short*)carve(32768ull * 2);
    unsigned short* Ur1T = (unsigned short*)carve(16384ull * 2);
    unsigned short* Ur2T = (unsigned short*)carve(16384ull * 2);

    // ---- input conversion / weight transpose (one-shot, tiny) ----
    k_cvt<<<36, 256, 0, stream>>>(pos_emb, PembB, 36 * 256);
    k_cvt<<<1, 256, 0, stream>>>(root_emb, RembB, 256);
    k_cvt<<<2, 256, 0, stream>>>(dec_emb, DembB, 512);
    k_cvt_t<<<256, 256, 0, stream>>>(Wp, WpT, 256, 256, 1);
    k_cvt_t<<<256, 256, 0, stream>>>(Ws, WsT, 256, 256, 1);
    k_cvt_t<<<256, 256, 0, stream>>>(Wc, WcT, 256, 256, 1);
    k_cvt_t<<<256, 256, 0, stream>>>(Wr, WrT, 256, 256, 1);
    k_cvt_t<<<256, 256, 0, stream>>>(Wd, WdT, 256, 256, 1);
    k_cvt_t<<<512, 256, 0, stream>>>(Wenc, WencT, 256, 256, 4);
    k_cvt_t<<<128, 256, 0, stream>>>(Ua1, Ua1T, 256, 128, 1);
    k_cvt_t<<<128, 256, 0, stream>>>(Ua2, Ua2T, 256, 128, 1);
    k_cvt_t<<<128, 256, 0, stream>>>(Ua3, Ua3T, 256, 128, 1);
    k_cvt_t<<<128, 256, 0, stream>>>(Ud1, Ud1T, 256, 128, 1);
    k_cvt_t<<<128, 256, 0, stream>>>(Ud2, Ud2T, 256, 128, 1);
    k_cvt_t<<<128, 256, 0, stream>>>(Ud3, Ud3T, 256, 128, 1);
    k_cvt_t<<<64, 256, 0, stream>>>(Ur1, Ur1T, 256, 64, 1);
    k_cvt_t<<<64, 256, 0, stream>>>(Ur2, Ur2T, 256, 64, 1);

    k_gather<<<4096, 256, 0, stream>>>(pos_tag, pos_emb, X);
    k_nosib<<<64, 256, 0, stream>>>(pos_emb, Yall);
    k_zero16<<<(4 * 48 * 128 + 255) / 256, 256, 0, stream>>>(FCb, 4 * 48 * 128);

    // y = x@W + b  (pre-encoder features, into Yall segments)
    launch_gemm(stream, X, 0, WpT, 0, bp, 0, nullptr, 0, Yall, 1, 0, 4096, 256, 256, 0, 0, 1);
    launch_gemm(stream, X, 0, WsT, 0, bs, 0, nullptr, 0, Yall + (size_t)4096 * 256, 1, 0, 4096, 256, 256, 0, 1, 1);
    launch_gemm(stream, PembB + 256, 0, WcT, 0, bc, 0, nullptr, 0, Yall + (size_t)8256 * 256, 1, 0, 34, 256, 256, 0, 0, 1);
    launch_gemm(stream, RembB, 0, WrT, 0, br, 0, nullptr, 0, Yall + (size_t)8290 * 256, 1, 0, 1, 256, 256, 0, 0, 1);
    launch_gemm(stream, DembB, 0, WdT, 0, bd, 0, nullptr, 0, Yall + (size_t)8291 * 256, 1, 0, 2, 256, 256, 0, 0, 1);

    // SkipConnect encoder: Hall[dv] = y + relu(y@W_enc[dv] + b_enc[dv])
    launch_gemm(stream, Yall, 0, WencT, 65536, benc, 256, Yall, 0,
                Hall, 1, 8293ll * 256, 8293, 256, 256, 1, 0, 4);

    const long long HDV = 8293ll * 256;
    // factor projections
    launch_gemm(stream, Hall, HDV, Ua1T, 0, nullptr, 0, nullptr, 0, FPb, 1, 4096ll * 128, 4096, 128, 256, 0, 0, 4);
    launch_gemm(stream, Hall, HDV, Ud1T, 0, nullptr, 0, nullptr, 0, GPb, 1, 4096ll * 128, 4096, 128, 256, 0, 0, 4);
    launch_gemm(stream, Hall + (size_t)4096 * 256, HDV, Ua3T, 0, nullptr, 0, nullptr, 0, FSb, 1, 4160ll * 128, 4160, 128, 256, 0, 0, 4);
    launch_gemm(stream, Hall + (size_t)4096 * 256, HDV, Ud2T, 0, nullptr, 0, nullptr, 0, GSb, 1, 4160ll * 128, 4160, 128, 256, 0, 0, 4);
    launch_gemm(stream, Hall + (size_t)8256 * 256, HDV, Ua2T, 0, nullptr, 0, nullptr, 0, FCb, 1, 48ll * 128, 34, 128, 256, 0, 0, 4);
    launch_gemm(stream, Hall + (size_t)8291 * 256, HDV, Ud3T, 0, nullptr, 0, nullptr, 0, GDb, 1, 2ll * 128, 2, 128, 256, 0, 0, 4);
    launch_gemm(stream, Hall + (size_t)8290 * 256, HDV, Ur1T, 0, nullptr, 0, nullptr, 0, RRb, 1, 64ll, 1, 64, 256, 0, 0, 4);
    launch_gemm(stream, Hall + (size_t)8256 * 256, HDV, Ur2T, 0, nullptr, 0, nullptr, 0, RCb, 1, 34ll * 64, 34, 64, 256, 0, 0, 4);

    float* out = (float*)d_out;
    k_attach  <<<4096, 256, 0, stream>>>(FPb, FSb, FCb, pos_tag, out);
    k_decision<<<4096, 256, 0, stream>>>(GPb, GSb, GDb, out + 34078720ll);
    k_root    <<<1,    256, 0, stream>>>(RRb, RCb, pos_tag, out + 36208640ll);
}
